// walkGenerateNet_81527069213315
// MI455X (gfx1250) — compile-verified, matrix operand order checked by reading, not run on hardware
//
#include <hip/hip_runtime.h>
#include <hip/hip_bf16.h>

// ---------- problem constants ----------
#define BB   1024
#define TT   85
#define DIN  20
#define HH   1024
#define CC   27
#define OBJ  (TT*36)          // 3060
#define OBJP 3072             // K padded to 32
#define STEPS (TT-1)          // 84

typedef __bf16 v16bf __attribute__((ext_vector_type(16)));
typedef float  v8f   __attribute__((ext_vector_type(8)));

__device__ __forceinline__ v8f wmma_bf16(v16bf a, v16bf b, v8f c) {
  // (neg_a, A, neg_b, B, c_mod, C, reuse_a, reuse_b)
  return __builtin_amdgcn_wmma_f32_16x16x32_bf16(false, a, false, b, (short)0, c, false, false);
}

// K-position inside a 32-wide bf16 fragment tile for vector element e and lane-half h
// (ISA 7.12.2: 16-bit A 16x32 layout; B 32x16 mirrors it with N across lanes)
__device__ __forceinline__ int frag_k(int e, int halfv) {
  int v = e >> 1, slot = e & 1;
  return ((v >= 4) ? 16 : 0) + 8 * halfv + 2 * (v & 3) + slot;
}

// ---------- pack fp32 row-major A [M x lda] -> bf16 fragment buffer [mt][kt][lane][16] ----------
__global__ void pack_a_kernel(const float* __restrict__ A, __bf16* __restrict__ out,
                              int Mt, int Kt, int realK, int lda) {
  int gid  = blockIdx.x * blockDim.x + threadIdx.x;
  int lane = gid & 31;
  int tile = gid >> 5;
  if (tile >= Mt * Kt) return;
  int mt = tile / Kt, kt = tile % Kt;
  int m = mt * 16 + (lane & 15);
  int halfv = lane >> 4;
  v16bf vv;
  for (int e = 0; e < 16; ++e) {
    int k = kt * 32 + frag_k(e, halfv);
    float val = (k < realK) ? A[(size_t)m * lda + k] : 0.0f;
    vv[e] = (__bf16)val;
  }
  *(v16bf*)(out + (size_t)tile * 512 + lane * 16) = vv;
}

// ---------- pack fp32 row-major W [K x ldw] -> bf16 fragment buffer [nt][kt][lane][16] ----------
__global__ void pack_b_kernel(const float* __restrict__ W, __bf16* __restrict__ out,
                              int Nt, int Kt, int realK, int realN, int ldw) {
  int gid  = blockIdx.x * blockDim.x + threadIdx.x;
  int lane = gid & 31;
  int tile = gid >> 5;
  if (tile >= Nt * Kt) return;
  int nt = tile / Kt, kt = tile % Kt;
  int n = nt * 16 + (lane & 15);
  int halfv = lane >> 4;
  v16bf vv;
  for (int e = 0; e < 16; ++e) {
    int k = kt * 32 + frag_k(e, halfv);
    float val = (k < realK && n < realN) ? W[(size_t)k * ldw + n] : 0.0f;
    vv[e] = (__bf16)val;
  }
  *(v16bf*)(out + (size_t)tile * 512 + lane * 16) = vv;
}

// ---------- pack x time slices into A fragments, one 16x32 k-tile per (t, mtile) ----------
// step 0 uses x[:,0,0:20] at k=0..19; steps t>=1 use x[:,t,1:20] at k=1..19 (k=0 slot zero,
// its contribution is supplied by the serial rank-1 update prev * eW1[0,:]).
__global__ void pack_x_kernel(const float* __restrict__ x, __bf16* __restrict__ out) {
  int gid  = blockIdx.x * blockDim.x + threadIdx.x;
  int lane = gid & 31;
  int mt   = (gid >> 5) & 63;
  int t    = gid >> 11;           // 64 mtiles * 32 lanes per t
  if (t >= STEPS) return;
  int b = mt * 16 + (lane & 15);
  int halfv = lane >> 4;
  v16bf vv;
  for (int e = 0; e < 16; ++e) {
    int k = frag_k(e, halfv);
    float val = 0.0f;
    if (t == 0) { if (k < DIN) val = x[((size_t)b * TT + 0) * DIN + k]; }
    else        { if (k >= 1 && k < DIN) val = x[((size_t)b * TT + t) * DIN + k]; }
    vv[e] = (__bf16)val;
  }
  *(v16bf*)(out + ((size_t)t * 64 + mt) * 512 + lane * 16) = vv;
}

// ---------- generic bf16 WMMA GEMM, 32x32 tile per wave (2x2 wmma) ----------
// Ping-pong double-buffered fragments: each half-iteration issues the other
// buffer's loads before its own WMMAs, so the load-wait overlaps WMMA latency
// without any register-rotation copies. C = optrelu(Ap @ Bp + bias), fp32 out.
__global__ void gemm_frag_bf16(const __bf16* __restrict__ Ap, const __bf16* __restrict__ Bp,
                               const float* __restrict__ bias, float* __restrict__ Cout,
                               int Mt, int Nt, int Kt, int relu) {
  int wid  = (blockIdx.x * blockDim.x + threadIdx.x) >> 5;   // wave-uniform
  int lane = threadIdx.x & 31;
  int Mt2 = Mt >> 1, Nt2 = Nt >> 1;
  if (wid >= Mt2 * Nt2) return;
  int mt2 = wid / Nt2, nt2 = wid % Nt2;

  const __bf16* a0 = Ap + (size_t)(mt2 * 2 + 0) * Kt * 512 + lane * 16;
  const __bf16* a1 = Ap + (size_t)(mt2 * 2 + 1) * Kt * 512 + lane * 16;
  const __bf16* b0 = Bp + (size_t)(nt2 * 2 + 0) * Kt * 512 + lane * 16;
  const __bf16* b1 = Bp + (size_t)(nt2 * 2 + 1) * Kt * 512 + lane * 16;

  v8f acc00 = {}, acc01 = {}, acc10 = {}, acc11 = {};

  // buffer F holds k, buffer G holds k+1
  v16bf fa0 = *(const v16bf*)a0, fa1 = *(const v16bf*)a1;
  v16bf fb0 = *(const v16bf*)b0, fb1 = *(const v16bf*)b1;

  int k = 0;
  for (;;) {
    v16bf ga0, ga1, gb0, gb1;
    bool have1 = (k + 1 < Kt);                       // uniform (Kt even in all uses)
    if (have1) {
      size_t off = (size_t)(k + 1) * 512;
      ga0 = *(const v16bf*)(a0 + off);
      ga1 = *(const v16bf*)(a1 + off);
      gb0 = *(const v16bf*)(b0 + off);
      gb1 = *(const v16bf*)(b1 + off);
      if (k + 8 < Kt) {
        size_t poff = (size_t)(k + 8) * 512;
        __builtin_prefetch(a0 + poff, 0, 1);
        __builtin_prefetch(a1 + poff, 0, 1);
        __builtin_prefetch(b0 + poff, 0, 1);
        __builtin_prefetch(b1 + poff, 0, 1);
      }
    }
    acc00 = wmma_bf16(fa0, fb0, acc00);
    acc01 = wmma_bf16(fa0, fb1, acc01);
    acc10 = wmma_bf16(fa1, fb0, acc10);
    acc11 = wmma_bf16(fa1, fb1, acc11);
    if (!have1) break;

    bool have2 = (k + 2 < Kt);
    if (have2) {
      size_t off = (size_t)(k + 2) * 512;
      fa0 = *(const v16bf*)(a0 + off);
      fa1 = *(const v16bf*)(a1 + off);
      fb0 = *(const v16bf*)(b0 + off);
      fb1 = *(const v16bf*)(b1 + off);
    }
    acc00 = wmma_bf16(ga0, gb0, acc00);
    acc01 = wmma_bf16(ga0, gb1, acc01);
    acc10 = wmma_bf16(ga1, gb0, acc10);
    acc11 = wmma_bf16(ga1, gb1, acc11);
    if (!have2) break;
    k += 2;
  }

  int halfv = lane >> 4;
  int N = Nt * 16;
  for (int i = 0; i < 2; ++i) {
    for (int j = 0; j < 2; ++j) {
      v8f acc = (i == 0) ? (j == 0 ? acc00 : acc01) : (j == 0 ? acc10 : acc11);
      int col = (nt2 * 2 + j) * 16 + (lane & 15);
      float bv = bias[col];
      for (int r = 0; r < 8; ++r) {
        int row = (mt2 * 2 + i) * 16 + r + 8 * halfv;  // C/D layout: VGPR r -> M=r (+8 upper half)
        float v = acc[r] + bv;
        if (relu) v = fmaxf(v, 0.0f);
        Cout[(size_t)row * N + col] = v;
      }
    }
  }
}

// ---------- serial step: h = relu(Z + x_t@Wtop + prev*r0); out_t = h @ eW2 + eb2 ----------
// 128 threads = 4 waves; each wave owns 16 batch rows. Fused two-layer via LDS A-fragments.
__global__ void step_kernel(const __bf16* __restrict__ AxT,   // [64 mtiles][lane][16]
                            const __bf16* __restrict__ WtopP, // [64 nt][1 kt] frags
                            const float*  __restrict__ Z,     // [B x H] fp32
                            const float*  __restrict__ r0,    // eW1 row 0 (H floats)
                            const float*  __restrict__ prevIn,
                            const __bf16* __restrict__ W2p,   // [2 nt][32 kt] frags
                            const float*  __restrict__ eb2,
                            float* __restrict__ out,          // [B][STEPS][C]
                            float* __restrict__ prevOut,
                            int t) {
  __shared__ alignas(32) __bf16 hLds[4][8][512];   // 32 KB: per-wave A-frag staging
  int wave = threadIdx.x >> 5, lane = threadIdx.x & 31;
  int mt = blockIdx.x * 4 + wave;                  // 16-row batch tile (0..63)
  int mbase = mt * 16;
  int halfv = lane >> 4;
  int ncol = lane & 15;

  v16bf afx = *(const v16bf*)(AxT + (size_t)mt * 512 + lane * 16);

  // hoist the serial-recurrence scalars off the inner loop (8 rows owned by this lane)
  float prevRow[8];
  for (int r = 0; r < 8; ++r)
    prevRow[r] = (t > 0) ? prevIn[mbase + r + 8 * halfv] : 0.0f;

  v8f acc0 = {}, acc1 = {};

  for (int phase = 0; phase < 4; ++phase) {
    // ---- stage 1: hidden columns [phase*256, phase*256+256) ----
    for (int jj = 0; jj < 16; ++jj) {
      int jt = phase * 16 + jj;
      v16bf bfx = *(const v16bf*)(WtopP + (size_t)jt * 512 + lane * 16);
      v8f acc = {};
      acc = wmma_bf16(afx, bfx, acc);
      int col = jt * 16 + ncol;
      float r0v = r0[col];
      int ktl = (col >> 5) & 7;                    // k-tile index inside this phase
      int e   = 8 * ((col >> 4) & 1) + (col & 7);  // invert fragment K mapping
      int h2  = (col >> 3) & 1;
      for (int r = 0; r < 8; ++r) {
        int row = mbase + r + 8 * halfv;
        float v = acc[r] + Z[(size_t)row * HH + col] + prevRow[r] * r0v;
        v = fmaxf(v, 0.0f);
        int dlane = (r + 8 * halfv) + 16 * h2;     // A-frag lane for (row,k)
        hLds[wave][ktl][dlane * 16 + e] = (__bf16)v;
      }
    }
    // ---- stage 2: accumulate output GEMM over this phase's 8 k-tiles ----
    for (int kk = 0; kk < 8; ++kk) {
      int kt = phase * 8 + kk;
      v16bf af = *(const v16bf*)(&hLds[wave][kk][lane * 16]);
      v16bf b0 = *(const v16bf*)(W2p + (size_t)(0 * 32 + kt) * 512 + lane * 16);
      v16bf b1 = *(const v16bf*)(W2p + (size_t)(1 * 32 + kt) * 512 + lane * 16);
      acc0 = wmma_bf16(af, b0, acc0);
      acc1 = wmma_bf16(af, b1, acc1);
    }
  }

  // ---- epilogue: bias, write out[:, t, :], export channel 0 for the recurrence ----
  for (int nt2 = 0; nt2 < 2; ++nt2) {
    v8f acc = nt2 ? acc1 : acc0;
    int c = nt2 * 16 + ncol;
    if (c < CC) {
      float bv = eb2[c];
      for (int r = 0; r < 8; ++r) {
        int row = mbase + r + 8 * halfv;
        float v = acc[r] + bv;
        out[(size_t)row * (STEPS * CC) + (size_t)t * CC + c] = v;
        if (c == 0) prevOut[row] = v;
      }
    }
  }
}

// ---------- host orchestration ----------
extern "C" void kernel_launch(void* const* d_in, const int* in_sizes, int n_in,
                              void* d_out, int out_size, void* d_ws, size_t ws_size,
                              hipStream_t stream) {
  const float* o   = (const float*)d_in[0];
  const float* x   = (const float*)d_in[1];
  const float* oW1 = (const float*)d_in[2];
  const float* ob1 = (const float*)d_in[3];
  const float* oW2 = (const float*)d_in[4];
  const float* ob2 = (const float*)d_in[5];
  const float* eW1 = (const float*)d_in[6];
  const float* eb1 = (const float*)d_in[7];
  const float* eW2 = (const float*)d_in[8];
  const float* eb2 = (const float*)d_in[9];
  float* out = (float*)d_out;
  char*  ws  = (char*)d_ws;

  // workspace layout (bytes)
  const size_t OFF_AP1  = 0;                          // 1024x3072 bf16
  const size_t OFF_WP1  = OFF_AP1  + (size_t)BB   * OBJP * 2;
  const size_t OFF_H1   = OFF_WP1  + (size_t)OBJP * HH   * 2;   // fp32 1024x1024
  const size_t OFF_AP2  = OFF_H1   + (size_t)BB   * HH   * 4;
  const size_t OFF_WP2  = OFF_AP2  + (size_t)BB   * HH   * 2;
  const size_t OFF_OI   = OFF_WP2  + (size_t)HH   * HH   * 2;   // objInfo fp32
  const size_t OFF_AP3  = OFF_OI   + (size_t)BB   * HH   * 4;
  const size_t OFF_WP3  = OFF_AP3  + (size_t)BB   * HH   * 2;
  const size_t OFF_Z    = OFF_WP3  + (size_t)HH   * HH   * 2;   // Z fp32
  const size_t OFF_WTOP = OFF_Z    + (size_t)BB   * HH   * 4;   // 32x1024 frags
  const size_t OFF_W2P  = OFF_WTOP + (size_t)32   * HH   * 2;   // 1024x32 frags
  const size_t OFF_AX   = OFF_W2P  + (size_t)HH   * 32   * 2;   // 84 x 1024x32 frags
  const size_t OFF_PRV0 = OFF_AX   + (size_t)STEPS * BB  * 32 * 2;
  const size_t OFF_PRV1 = OFF_PRV0 + (size_t)BB * 4;
  const size_t NEED     = OFF_PRV1 + (size_t)BB * 4;
  if (ws_size < NEED) return;

  __bf16* Ap1  = (__bf16*)(ws + OFF_AP1);
  __bf16* Wp1  = (__bf16*)(ws + OFF_WP1);
  float*  H1   = (float*)(ws + OFF_H1);
  __bf16* Ap2  = (__bf16*)(ws + OFF_AP2);
  __bf16* Wp2  = (__bf16*)(ws + OFF_WP2);
  float*  OI   = (float*)(ws + OFF_OI);
  __bf16* Ap3  = (__bf16*)(ws + OFF_AP3);
  __bf16* Wp3  = (__bf16*)(ws + OFF_WP3);
  float*  Zb   = (float*)(ws + OFF_Z);
  __bf16* WtopP= (__bf16*)(ws + OFF_WTOP);
  __bf16* W2p  = (__bf16*)(ws + OFF_W2P);
  __bf16* AxP  = (__bf16*)(ws + OFF_AX);
  float*  prevBuf[2] = { (float*)(ws + OFF_PRV0), (float*)(ws + OFF_PRV1) };

  auto blk = [](long threads) { return (int)((threads + 255) / 256); };
  const long gemmWaves = (long)32 * 32 * 32;   // (64/2)*(64/2) waves * 32 lanes

  // --- objInfo path: H1 = relu(o @ oW1 + ob1); OI = H1 @ oW2 + ob2 ---
  pack_a_kernel<<<blk((long)64*96*32), 256, 0, stream>>>(o,   Ap1, 64, 96, OBJ, OBJ);
  pack_b_kernel<<<blk((long)64*96*32), 256, 0, stream>>>(oW1, Wp1, 64, 96, OBJ, HH, HH);
  gemm_frag_bf16<<<blk(gemmWaves), 256, 0, stream>>>(Ap1, Wp1, ob1, H1, 64, 64, 96, 1);

  pack_a_kernel<<<blk((long)64*32*32), 256, 0, stream>>>(H1,  Ap2, 64, 32, HH, HH);
  pack_b_kernel<<<blk((long)64*32*32), 256, 0, stream>>>(oW2, Wp2, 64, 32, HH, HH, HH);
  gemm_frag_bf16<<<blk(gemmWaves), 256, 0, stream>>>(Ap2, Wp2, ob2, OI, 64, 64, 32, 0);

  // --- Z = objInfo @ eW1[20:,:] + eb1 (step-invariant hidden pre-activation) ---
  pack_a_kernel<<<blk((long)64*32*32), 256, 0, stream>>>(OI, Ap3, 64, 32, HH, HH);
  pack_b_kernel<<<blk((long)64*32*32), 256, 0, stream>>>(eW1 + (size_t)DIN * HH, Wp3, 64, 32, HH, HH, HH);
  gemm_frag_bf16<<<blk(gemmWaves), 256, 0, stream>>>(Ap3, Wp3, eb1, Zb, 64, 64, 32, 0);

  // --- small packed weights for the step loop ---
  pack_b_kernel<<<blk((long)64*1*32),  256, 0, stream>>>(eW1, WtopP, 64, 1, DIN, HH, HH);
  pack_b_kernel<<<blk((long)2*32*32),  256, 0, stream>>>(eW2, W2p,   2, 32, HH, CC, CC);
  pack_x_kernel<<<blk((long)STEPS*64*32), 256, 0, stream>>>(x, AxP);

  // --- 84 dependent steps; only the rank-1 prev*r0 term is serial ---
  for (int t = 0; t < STEPS; ++t) {
    step_kernel<<<16, 128, 0, stream>>>(AxP + (size_t)t * 64 * 512, WtopP, Zb, eW1,
                                        prevBuf[t & 1], W2p, eb2, out,
                                        prevBuf[(t + 1) & 1], t);
  }
}
